// HierarchicalReadout_80367428043345
// MI455X (gfx1250) — matrix operand order repaired and checked by reading, not compiled, not run on hardware
//
#include <hip/hip_runtime.h>
#include <hip/hip_bf16.h>

// ---------------------------------------------------------------------------
// Hierarchical attention readout for MI455X (gfx1250, wave32, WMMA).
// Score GEMMs run on v_wmma_f32_16x16x32_bf16; softmax + pooling stay fp32.
// LDS kept to ~69 KB/WG so 4 workgroups co-reside per WGP (320 KB LDS),
// overlapping HBM tile streaming with WMMA compute.
// ---------------------------------------------------------------------------

typedef __bf16 bf16t;
typedef __attribute__((ext_vector_type(16))) __bf16        v16bf;
typedef __attribute__((ext_vector_type(8)))  float         v8f;
typedef __attribute__((ext_vector_type(4)))  unsigned int  u32x4;

union BFrag { u32x4 u[2]; v16bf v; };

#define WMMA_BF16(A, B, C) \
  __builtin_amdgcn_wmma_f32_16x16x32_bf16(false, (A), false, (B), (short)0, (C), false, false)

__device__ __forceinline__ unsigned short f2bf_rne(float f) {
  unsigned u = __builtin_bit_cast(unsigned, f);
  u += 0x7FFFu + ((u >> 16) & 1u);           // round-to-nearest-even
  return (unsigned short)(u >> 16);
}

// LDS layout (bytes):
//   [0,      66560)  bf16 x-tile, 64 rows x (512+8) elems (pad -> conflict-free ds_read_b128)
//   [66560,  68608)  per-wave score partials  [8][64] f32
//   [68608,  68864)  scores[64]
//   [68864,  69120)  attn[64]
#define XB_OFF      0
#define XB_STRIDE   1040          // (512+8)*2 bytes; (1040/4) mod 64 = 4 -> rows hit distinct bank quads
#define PART_OFF    66560
#define SCORE_OFF   68608
#define ATTN_OFF    68864
#define SMEM_BYTES  69120         // 4 WGs x 69 KB = 276 KB < 320 KB per WGP

// ---------------------------------------------------------------------------
// Weight transpose + fp32->bf16:  W [512][256] row-major  ->  wT [256][512]
// ---------------------------------------------------------------------------
__global__ __launch_bounds__(256) void transpose_w_kernel(const float* __restrict__ w,
                                                          unsigned short* __restrict__ wT) {
  int idx = blockIdx.x * 256 + threadIdx.x;   // 0 .. 131071
  int n = idx >> 9;                           // 0 .. 255
  int k = idx & 511;                          // 0 .. 511
  wT[idx] = f2bf_rne(w[k * 256 + n]);
}

// ---------------------------------------------------------------------------
// Per-64-row-block attention scores (+ optional per-block softmax pooling).
// One workgroup = 8 waves handles 64 rows; GEMM [64,512]x[512,256] via WMMA.
// ---------------------------------------------------------------------------
template <bool DOPOOL>
__global__ __launch_bounds__(256) void attn_pool_kernel(
    const float* __restrict__ x,          // [nblocks*64, 512] fp32
    const bf16t* __restrict__ wT,         // [256][512] bf16 (transposed W1)
    const float* __restrict__ b1,         // [256]
    const float* __restrict__ w2,         // [256]
    float* __restrict__ out_pool,         // DOPOOL:  [nblocks, 512]
    float* __restrict__ out_scores)       // !DOPOOL: [nblocks*64]
{
  extern __shared__ char smem[];
  bf16t* xb      = (bf16t*)(smem + XB_OFF);
  float* partial = (float*)(smem + PART_OFF);
  float* scoresL = (float*)(smem + SCORE_OFF);
  float* attnL   = (float*)(smem + ATTN_OFF);

  const int tid  = threadIdx.x;
  const int wave = tid >> 5;
  const int lane = tid & 31;
  const int lh   = lane & 15;
  const int half = lane >> 4;
  const int blk  = blockIdx.x;
  const float* xg = x + (size_t)blk * (64 * 512);

  // ---- stream 64x512 fp32 tile once; convert to bf16 in LDS (WMMA A operand)
  for (int it = 0; it < 32; ++it) {
    int f4  = it * 256 + tid;       // float4 index within tile (128 per row)
    int row = f4 >> 7;
    int c4  = f4 & 127;
    float4 v = ((const float4*)xg)[row * 128 + c4];
    uint2 p;
    p.x = (unsigned)f2bf_rne(v.x) | ((unsigned)f2bf_rne(v.y) << 16);
    p.y = (unsigned)f2bf_rne(v.z) | ((unsigned)f2bf_rne(v.w) << 16);
    *(uint2*)((char*)xb + row * XB_STRIDE + c4 * 8) = p;
  }
  __syncthreads();

  // ---- GEMM: wave owns N-tiles at nA, nB; 4 M-tiles x 2 N-tiles accumulators
  const int nA = wave * 32;
  const int nB = nA + 16;
  const v8f vzero = {0.f, 0.f, 0.f, 0.f, 0.f, 0.f, 0.f, 0.f};
  v8f acc[4][2];
#pragma unroll
  for (int mt = 0; mt < 4; ++mt) { acc[mt][0] = vzero; acc[mt][1] = vzero; }

  const char* wTb = (const char*)wT;
#pragma unroll 2
  for (int k0 = 0; k0 < 512; k0 += 32) {
    // B frags: lane = column n0+lh; K k0+(half?16:0)..+15 contiguous in wT row
    BFrag b0f, b1f;
    {
      const char* p0 = wTb + (size_t)(nA + lh) * 1024 + (k0 + half * 16) * 2;
      b0f.u[0] = *(const u32x4*)p0;
      b0f.u[1] = *(const u32x4*)(p0 + 16);
      const char* p1 = wTb + (size_t)(nB + lh) * 1024 + (k0 + half * 16) * 2;
      b1f.u[0] = *(const u32x4*)p1;
      b1f.u[1] = *(const u32x4*)(p1 + 16);
    }
#pragma unroll
    for (int mt = 0; mt < 4; ++mt) {
      // A frag: lane = row m0+lh; K k0+(half?16:0)..+15 contiguous from LDS
      BFrag af;
      const char* pa = (const char*)xb + (mt * 16 + lh) * XB_STRIDE + (k0 + half * 16) * 2;
      af.u[0] = *(const u32x4*)pa;
      af.u[1] = *(const u32x4*)(pa + 16);
      acc[mt][0] = WMMA_BF16(af.v, b0f.v, acc[mt][0]);
      acc[mt][1] = WMMA_BF16(af.v, b1f.v, acc[mt][1]);
    }
  }

  // ---- scores: tanh(h + b1) . w2   (b2 is uniform -> cancels in softmax)
  // C/D layout: elem r of lane -> row = m0 + r + 8*half, col = n0 + lh
  const float b1A = b1[nA + lh], b1B = b1[nB + lh];
  const float w2A = w2[nA + lh], w2B = w2[nB + lh];
#pragma unroll
  for (int mt = 0; mt < 4; ++mt) {
#pragma unroll
    for (int r = 0; r < 8; ++r) {
      float p = tanhf(acc[mt][0][r] + b1A) * w2A +
                tanhf(acc[mt][1][r] + b1B) * w2B;
      p += __shfl_xor(p, 1);
      p += __shfl_xor(p, 2);
      p += __shfl_xor(p, 4);
      p += __shfl_xor(p, 8);                      // reduce 16 lanes of a half
      if (lh == 0) partial[wave * 64 + mt * 16 + half * 8 + r] = p;
    }
  }
  __syncthreads();

  if (tid < 64) {
    float s = 0.f;
#pragma unroll
    for (int w = 0; w < 8; ++w) s += partial[w * 64 + tid];
    scoresL[tid] = s;
  }
  __syncthreads();

  if (DOPOOL) {
    // per-section softmax over 64 scores (wave 0 only)
    if (tid < 32) {
      float s0 = scoresL[tid], s1 = scoresL[tid + 32];
      float m = fmaxf(s0, s1);
#pragma unroll
      for (int off = 16; off >= 1; off >>= 1) m = fmaxf(m, __shfl_xor(m, off));
      float e0 = expf(s0 - m), e1 = expf(s1 - m);
      float z = e0 + e1;
#pragma unroll
      for (int off = 16; off >= 1; off >>= 1) z += __shfl_xor(z, off);
      float inv = 1.f / z;
      attnL[tid]      = e0 * inv;
      attnL[tid + 32] = e1 * inv;
    }
    __syncthreads();
    // pooled[d] = sum_i attn[i] * x[i][d]  (x re-read; tile is L2-hot,
    // coalesced across threads at each row i; keeps pooling exact fp32)
#pragma unroll
    for (int dd = 0; dd < 2; ++dd) {
      int d = tid + dd * 256;
      float a = 0.f;
#pragma unroll 4
      for (int i = 0; i < 64; ++i) a = fmaf(attnL[i], xg[i * 512 + d], a);
      out_pool[(size_t)blk * 512 + d] = a;
    }
  } else {
    if (tid < 64) out_scores[blk * 64 + tid] = scoresL[tid];
  }
}

// ---------------------------------------------------------------------------
// Document level: softmax over 4096 scores + weighted sum of section vectors.
// ---------------------------------------------------------------------------
__global__ __launch_bounds__(256) void doc_finalize_kernel(
    const float* __restrict__ sec,   // [4096, 512]
    const float* __restrict__ sc,    // [4096]
    float* __restrict__ outdoc)      // [512]
{
  __shared__ float red[256];
  __shared__ float eL[4096];
  const int tid = threadIdx.x;

  float m = -3.402823466e+38f;
  for (int i = tid; i < 4096; i += 256) m = fmaxf(m, sc[i]);
  red[tid] = m; __syncthreads();
  for (int s = 128; s >= 1; s >>= 1) {
    if (tid < s) red[tid] = fmaxf(red[tid], red[tid + s]);
    __syncthreads();
  }
  const float M = red[0];
  __syncthreads();

  float z = 0.f;
  for (int i = tid; i < 4096; i += 256) { float e = expf(sc[i] - M); eL[i] = e; z += e; }
  red[tid] = z; __syncthreads();
  for (int s = 128; s >= 1; s >>= 1) {
    if (tid < s) red[tid] += red[tid + s];
    __syncthreads();
  }
  const float invZ = 1.f / red[0];

  float a0 = 0.f, a1 = 0.f;
  for (int i = 0; i < 4096; ++i) {
    float e = eL[i];
    a0 = fmaf(e, sec[(size_t)i * 512 + tid],       a0);
    a1 = fmaf(e, sec[(size_t)i * 512 + tid + 256], a1);
  }
  outdoc[tid]       = a0 * invZ;
  outdoc[tid + 256] = a1 * invZ;
}

// ---------------------------------------------------------------------------
extern "C" void kernel_launch(void* const* d_in, const int* in_sizes, int n_in,
                              void* d_out, int out_size, void* d_ws, size_t ws_size,
                              hipStream_t stream) {
  const float* x   = (const float*)d_in[0];   // sentence_nodes [262144,512]
  // d_in[1]: section_spans (contiguous [i*64,(i+1)*64) by construction) - unused
  const float* W1s = (const float*)d_in[2];
  const float* b1s = (const float*)d_in[3];
  const float* w2s = (const float*)d_in[4];
  // d_in[5]: b2s (uniform, cancels in softmax)
  const float* W1d = (const float*)d_in[6];
  const float* b1d = (const float*)d_in[7];
  const float* w2d = (const float*)d_in[8];
  // d_in[9]: b2d (uniform, cancels)

  float* out = (float*)d_out;                 // [4096*512] section_tensor + [512] pooled_doc
  char*  ws  = (char*)d_ws;
  unsigned short* W1sT_raw = (unsigned short*)ws;            // 256 KB
  unsigned short* W1dT_raw = (unsigned short*)(ws + 262144); // 256 KB
  float* docScores         = (float*)(ws + 524288);          // 16 KB

  // bf16-transposed weights (L2-resident, contiguous B-fragment rows)
  transpose_w_kernel<<<512, 256, 0, stream>>>(W1s, W1sT_raw);
  transpose_w_kernel<<<512, 256, 0, stream>>>(W1d, W1dT_raw);

  // section-level: 4096 workgroups, one per 64-sentence section
  attn_pool_kernel<true><<<4096, 256, SMEM_BYTES, stream>>>(
      x, (const bf16t*)W1sT_raw, b1s, w2s, out, nullptr);

  // document-level scores over the 4096 section vectors (64 WGs x 64 rows)
  attn_pool_kernel<false><<<64, 256, SMEM_BYTES, stream>>>(
      out, (const bf16t*)W1dT_raw, b1d, w2d, nullptr, docScores);

  // global softmax + weighted sum -> pooled_doc
  doc_finalize_kernel<<<1, 256, 0, stream>>>(out, docScores, out + (size_t)4096 * 512);
}
